// Attention_Rel_Scl_3719441679010
// MI455X (gfx1250) — matrix operand order repaired and checked by reading, not compiled, not run on hardware
//
#include <hip/hip_runtime.h>

// ---------------------------------------------------------------------------
// Types for CDNA5 WMMA (wave32): D(16x16 f32) = A(16x32 bf16) x B(32x16 bf16) + C
// ---------------------------------------------------------------------------
typedef __attribute__((ext_vector_type(16))) __bf16 v16bf;
typedef __attribute__((ext_vector_type(8)))  __bf16 v8bf;
typedef __attribute__((ext_vector_type(8)))  float  v8f;

#define SHUF16(lo, hi) __builtin_shufflevector(lo, hi, 0,1,2,3,4,5,6,7,8,9,10,11,12,13,14,15)
#define WMMA_BF16(a, b, c) \
    __builtin_amdgcn_wmma_f32_16x16x32_bf16(false, (a), false, (b), (short)0, (c), false, false)

__device__ __forceinline__ unsigned short f2bf(float f) {
    union { float f; unsigned u; } v; v.f = f;
    unsigned r = v.u + 0x7FFFu + ((v.u >> 16) & 1u);   // round-to-nearest-even
    return (unsigned short)(r >> 16);
}

// A-fragment (16x32 bf16, row-major source, K contiguous).
// Lane l holds row (l&15): elems 0..7 = K[b0..b0+7], elems 8..15 = K[b0+16..b0+23],
// b0 = (l>=16) ? 8 : 0.   p = row base + k-offset.
__device__ __forceinline__ v16bf ldA(const __bf16* p, int lane) {
    const int b0 = (lane >> 4) << 3;
    v8bf lo = *(const v8bf*)(p + b0);
    v8bf hi = *(const v8bf*)(p + b0 + 16);
    return SHUF16(lo, hi);
}
// B-fragment (32x16 bf16). Lane l holds column (l&15): elems 0..7 = K[b0..b0+7], elems 8..15 = K[b0+16..b0+23],
// b0 = (l>=16) ? 8 : 0.   p = column base (K contiguous) + k-offset.
__device__ __forceinline__ v16bf ldB(const __bf16* p, int lane) {
    const int b0 = (lane >> 4) << 3;
    v8bf lo = *(const v8bf*)(p + b0);
    v8bf hi = *(const v8bf*)(p + b0 + 16);
    return SHUF16(lo, hi);
}

// ---------------------------------------------------------------------------
// Kernel 0: fp32 -> bf16 convert
// ---------------------------------------------------------------------------
__global__ __launch_bounds__(256) void cvt_bf16_kernel(const float* __restrict__ src,
                                                       unsigned short* __restrict__ dst,
                                                       int n) {
    int i = blockIdx.x * 256 + threadIdx.x;
    if (i < n) dst[i] = f2bf(src[i]);
}

// ---------------------------------------------------------------------------
// Kernel 1: fused QKV projection.  y = x @ W^T  (M=4096, N=1024, K=1024)
// blockIdx.z selects Q / K / V.  Block tile 64x128, 8 waves, wave tile 16x64.
// k-loop is software-pipelined (double-buffered fragments) so WMMA overlaps
// the next step's global loads instead of draining loadcnt to 0 every WMMA.
// Q,K stored [B,H,S,d] bf16; V stored transposed [B,H,d,S] bf16.
// ---------------------------------------------------------------------------
__global__ __launch_bounds__(256) void qkv_gemm_kernel(
    const __bf16* __restrict__ xb,
    const __bf16* __restrict__ wqb,
    const __bf16* __restrict__ wkb,
    const __bf16* __restrict__ wvb,
    unsigned short* __restrict__ Qb,
    unsigned short* __restrict__ Kb,
    unsigned short* __restrict__ Vtb) {
    const int E = 1024;
    const int tid  = threadIdx.x;
    const int lane = tid & 31;
    const int wave = tid >> 5;
    const int msub = wave & 3;        // 4 row sub-tiles of 16
    const int nsub = wave >> 2;       // 2 col sub-tiles of 64
    const int m_base = blockIdx.x * 64 + msub * 16;
    const int n_base = blockIdx.y * 128 + nsub * 64;
    const int mat = blockIdx.z;
    const __bf16* W = (mat == 0) ? wqb : (mat == 1) ? wkb : wvb;

    const int row  = m_base + (lane & 15);
    const __bf16* aRow = xb + row * E;
    const int ncol = n_base + (lane & 15);
    const __bf16* w0 = W + (ncol +  0) * E;
    const __bf16* w1 = W + (ncol + 16) * E;
    const __bf16* w2 = W + (ncol + 32) * E;
    const __bf16* w3 = W + (ncol + 48) * E;

    const v8f vzero = {0.f, 0.f, 0.f, 0.f, 0.f, 0.f, 0.f, 0.f};
    v8f acc0 = vzero, acc1 = vzero, acc2 = vzero, acc3 = vzero;

    // prologue: fragments for k = 0
    v16bf a  = ldA(aRow, lane);
    v16bf b0 = ldB(w0, lane);
    v16bf b1 = ldB(w1, lane);
    v16bf b2 = ldB(w2, lane);
    v16bf b3 = ldB(w3, lane);

    for (int k = 0; k < E - 32; k += 32) {
        const int kn = k + 32;
        __builtin_prefetch(aRow + kn + 256, 0, 1);          // global_prefetch
        v16bf an  = ldA(aRow + kn, lane);
        v16bf bn0 = ldB(w0 + kn, lane);
        v16bf bn1 = ldB(w1 + kn, lane);
        v16bf bn2 = ldB(w2 + kn, lane);
        v16bf bn3 = ldB(w3 + kn, lane);
        acc0 = WMMA_BF16(a, b0, acc0);
        acc1 = WMMA_BF16(a, b1, acc1);
        acc2 = WMMA_BF16(a, b2, acc2);
        acc3 = WMMA_BF16(a, b3, acc3);
        a = an; b0 = bn0; b1 = bn1; b2 = bn2; b3 = bn3;
    }
    // epilogue: last k step
    acc0 = WMMA_BF16(a, b0, acc0);
    acc1 = WMMA_BF16(a, b1, acc1);
    acc2 = WMMA_BF16(a, b2, acc2);
    acc3 = WMMA_BF16(a, b3, acc3);

    const int hi8 = (lane >> 4) << 3;   // C/D: VGPR r -> M = r + hi8, N = lane&15
    v8f accs[4] = {acc0, acc1, acc2, acc3};

    if (mat < 2) {
        unsigned short* dst = (mat == 0) ? Qb : Kb;
        for (int j = 0; j < 4; ++j) {
            const int n  = n_base + j * 16 + (lane & 15);
            const int h  = n >> 6, dd = n & 63;
            for (int r = 0; r < 8; ++r) {
                const int m = m_base + hi8 + r;
                const int bI = m >> 10, s = m & 1023;
                dst[(((bI * 16 + h) * 1024 + s) << 6) + dd] = f2bf(accs[j][r]);
            }
        }
    } else {
        // transposed store: 8 consecutive M values per lane pack into one b128
        const int m0 = m_base + hi8;
        const int bI = m0 >> 10, s0 = m0 & 1023;
        for (int j = 0; j < 4; ++j) {
            const int n = n_base + j * 16 + (lane & 15);
            const int h = n >> 6, dd = n & 63;
            unsigned short t[8];
            for (int r = 0; r < 8; ++r) t[r] = f2bf(accs[j][r]);
            uint4 pk;
            pk.x = t[0] | ((unsigned)t[1] << 16);
            pk.y = t[2] | ((unsigned)t[3] << 16);
            pk.z = t[4] | ((unsigned)t[5] << 16);
            pk.w = t[6] | ((unsigned)t[7] << 16);
            *(uint4*)(Vtb + (((bI * 16 + h) * 64 + dd) << 10) + s0) = pk;
        }
    }
}

// ---------------------------------------------------------------------------
// Kernel 2: attention for one (b, h, 32-row tile).
// Phase 1: scores(32x1024) = scale * Q Kt  -> LDS fp32 (128 KB of the 320 KB WGP LDS)
// Phase 2: row softmax (two-pass, fp32)
// Phase 3: p = softmax + rel_bias -> LDS bf16 (64 KB)
// Phase 4: out(32x64) = p @ V   (A from LDS via ds_read_b128, B from Vt)
// GEMM loops in phases 1 and 4 are software-pipelined.
// ---------------------------------------------------------------------------
__global__ __launch_bounds__(256) void attn_kernel(
    const __bf16* __restrict__ Qb,
    const __bf16* __restrict__ Kb,
    const __bf16* __restrict__ Vtb,
    const float* __restrict__ table,   // [2S-1, H]
    float* __restrict__ out) {         // [B, S, E] pre-LN
    const int S = 1024, H = 16;
    __shared__ float          s_sc[32 * 1024];   // fp32 scores
    __shared__ unsigned short s_p [32 * 1024];   // bf16 probabilities (+bias)
    __shared__ float          s_red[8 * 32];
    __shared__ float          s_rowmax[32];
    __shared__ float          s_rowinv[32];

    const int tid  = threadIdx.x;
    const int lane = tid & 31;
    const int wave = tid >> 5;
    const int st = blockIdx.x;           // 32-row tile index
    const int h  = blockIdx.y;
    const int bI = blockIdx.z;
    const int s0 = st * 32;
    const int hi8  = (lane >> 4) << 3;
    const int msub = wave & 1;           // 2 row sub-tiles of 16
    const float scale = 0.03125f;        // E^-0.5 = 1/32
    const v8f vzero = {0.f, 0.f, 0.f, 0.f, 0.f, 0.f, 0.f, 0.f};

    // ---- Phase 1: scores ----
    {
        const int quad = wave >> 1;      // column range quad*256 .. +256
        const __bf16* Qh = Qb + ((bI * H + h) * S) * 64;
        const __bf16* Kh = Kb + ((bI * H + h) * S) * 64;
        const __bf16* qptr = Qh + (s0 + msub * 16 + (lane & 15)) * 64;
        v16bf aq0 = ldA(qptr + 0,  lane);   // d = 0..31
        v16bf aq1 = ldA(qptr + 32, lane);   // d = 32..63

        const __bf16* kcol = Kh + (quad * 256 + (lane & 15)) * 64;
        v16bf b0 = ldB(kcol + 0,  lane);    // nt = 0 fragments
        v16bf b1 = ldB(kcol + 32, lane);
        for (int nt = 0; nt < 16; ++nt) {
            v16bf bn0, bn1;
            if (nt < 15) {                              // prefetch nt+1 fragments
                const __bf16* knext = kcol + (nt + 1) * 16 * 64;
                bn0 = ldB(knext + 0,  lane);
                bn1 = ldB(knext + 32, lane);
            }
            v8f acc = vzero;
            acc = WMMA_BF16(aq0, b0, acc);
            acc = WMMA_BF16(aq1, b1, acc);
            const int tcol = quad * 256 + nt * 16 + (lane & 15);
            for (int r = 0; r < 8; ++r)
                s_sc[(msub * 16 + hi8 + r) * 1024 + tcol] = acc[r] * scale;
            b0 = bn0; b1 = bn1;
        }
    }
    __syncthreads();

    // ---- Phase 2: softmax stats ----
    const int r   = tid & 31;            // row within tile
    const int seg = tid >> 5;            // 8 segments of 128 cols
    const float* srow = s_sc + r * 1024 + seg * 128;
    float mx = -3.4e38f;
    for (int c = 0; c < 128; ++c) mx = fmaxf(mx, srow[c]);
    s_red[seg * 32 + r] = mx;
    __syncthreads();
    if (tid < 32) {
        float mm = s_red[tid];
        for (int g = 1; g < 8; ++g) mm = fmaxf(mm, s_red[g * 32 + tid]);
        s_rowmax[tid] = mm;
    }
    __syncthreads();
    const float rm = s_rowmax[r];
    float sum = 0.f;
    for (int c = 0; c < 128; ++c) sum += __expf(srow[c] - rm);
    s_red[seg * 32 + r] = sum;
    __syncthreads();
    if (tid < 32) {
        float ss = 0.f;
        for (int g = 0; g < 8; ++g) ss += s_red[g * 32 + tid];
        s_rowinv[tid] = 1.0f / ss;
    }
    __syncthreads();

    // ---- Phase 3: p = softmax + relative bias (bias added AFTER softmax) ----
    {
        const float inv = s_rowinv[r];
        const int sg = s0 + r;
        const int cbase = seg * 128;
        for (int c = 0; c < 128; ++c) {
            const int t = cbase + c;
            const float bias = table[(sg - t + 1023) * H + h];
            const float p = __expf(s_sc[r * 1024 + t] - rm) * inv + bias;
            s_p[r * 1024 + t] = f2bf(p);
        }
    }
    __syncthreads();

    // ---- Phase 4: out = p @ V ----
    {
        const int nt = wave >> 1;                        // 4 d-tiles of 16
        const int prow = msub * 16 + (lane & 15);
        const int dd = nt * 16 + (lane & 15);
        const __bf16* vcol = Vtb + (((bI * H + h) * 64 + dd) << 10);
        const __bf16* pbase = (const __bf16*)s_p + prow * 1024;

        v8f acc = vzero;
        v16bf a = ldA(pbase, lane);                      // k = 0 fragments
        v16bf b = ldB(vcol, lane);
        for (int k = 0; k < 1024 - 32; k += 32) {
            v16bf an = ldA(pbase + k + 32, lane);        // ds_read_b128 x2
            v16bf bn = ldB(vcol + k + 32, lane);         // global_load_b128 x2
            acc = WMMA_BF16(a, b, acc);
            a = an; b = bn;
        }
        acc = WMMA_BF16(a, b, acc);

        for (int rr = 0; rr < 8; ++rr) {
            const int sgl = s0 + msub * 16 + hi8 + rr;
            out[((bI * 1024 + sgl) << 10) + h * 64 + dd] = acc[rr];
        }
    }
}

// ---------------------------------------------------------------------------
// Kernel 3: in-place LayerNorm over last dim (1024) of d_out.
// ---------------------------------------------------------------------------
__global__ __launch_bounds__(256) void ln_kernel(float* __restrict__ out,
                                                 const float* __restrict__ gamma,
                                                 const float* __restrict__ beta) {
    __shared__ float rs[256], rs2[256];
    const int row = blockIdx.x;
    const int tid = threadIdx.x;
    float* p = out + row * 1024;
    float v[4], s = 0.f, s2 = 0.f;
    for (int i = 0; i < 4; ++i) {
        v[i] = p[tid + i * 256];
        s += v[i];
        s2 += v[i] * v[i];
    }
    rs[tid] = s; rs2[tid] = s2;
    __syncthreads();
    for (int off = 128; off > 0; off >>= 1) {
        if (tid < off) { rs[tid] += rs[tid + off]; rs2[tid] += rs2[tid + off]; }
        __syncthreads();
    }
    const float mu  = rs[0] * (1.0f / 1024.0f);
    const float var = rs2[0] * (1.0f / 1024.0f) - mu * mu;
    const float rstd = rsqrtf(var + 1e-5f);
    for (int i = 0; i < 4; ++i) {
        const int c = tid + i * 256;
        p[c] = (v[i] - mu) * rstd * gamma[c] + beta[c];
    }
}

// ---------------------------------------------------------------------------
// Launch
// ---------------------------------------------------------------------------
extern "C" void kernel_launch(void* const* d_in, const int* in_sizes, int n_in,
                              void* d_out, int out_size, void* d_ws, size_t ws_size,
                              hipStream_t stream) {
    (void)in_sizes; (void)n_in; (void)out_size; (void)ws_size;
    const float* x     = (const float*)d_in[0];   // [4,1024,1024]
    const float* Wq    = (const float*)d_in[1];   // [1024,1024]
    const float* Wk    = (const float*)d_in[2];
    const float* Wv    = (const float*)d_in[3];
    const float* table = (const float*)d_in[4];   // [2047,16]
    const float* gamma = (const float*)d_in[5];
    const float* beta  = (const float*)d_in[6];
    float* out = (float*)d_out;

    char* ws = (char*)d_ws;
    // bf16 workspace layout (byte offsets)
    unsigned short* xb  = (unsigned short*)(ws + 0);         // 8 MB   x bf16
    unsigned short* wqb = (unsigned short*)(ws + (8u  << 20));
    unsigned short* wkb = (unsigned short*)(ws + (10u << 20));
    unsigned short* wvb = (unsigned short*)(ws + (12u << 20));
    unsigned short* Qb  = (unsigned short*)(ws + (14u << 20)); // [B,H,S,d]
    unsigned short* Kb  = (unsigned short*)(ws + (22u << 20)); // [B,H,S,d]
    unsigned short* Vtb = (unsigned short*)(ws + (30u << 20)); // [B,H,d,S]

    const int NX = 4 * 1024 * 1024, NW = 1024 * 1024;
    cvt_bf16_kernel<<<(NX + 255) / 256, 256, 0, stream>>>(x,  xb,  NX);
    cvt_bf16_kernel<<<(NW + 255) / 256, 256, 0, stream>>>(Wq, wqb, NW);
    cvt_bf16_kernel<<<(NW + 255) / 256, 256, 0, stream>>>(Wk, wkb, NW);
    cvt_bf16_kernel<<<(NW + 255) / 256, 256, 0, stream>>>(Wv, wvb, NW);

    qkv_gemm_kernel<<<dim3(64, 8, 3), 256, 0, stream>>>(
        (const __bf16*)xb, (const __bf16*)wqb, (const __bf16*)wkb,
        (const __bf16*)wvb, Qb, Kb, Vtb);

    attn_kernel<<<dim3(32, 16, 4), 256, 0, stream>>>(
        (const __bf16*)Qb, (const __bf16*)Kb, (const __bf16*)Vtb, table, out);

    ln_kernel<<<4096, 256, 0, stream>>>(out, gamma, beta);
}
